// Pedal_87256555585559
// MI455X (gfx1250) — matrix operand order repaired and checked by reading, not compiled, not run on hardware
//
#include <hip/hip_runtime.h>

#define PP 6
#define BB 4096
#define DD 256
#define NN 8192
#define KK 10
#define FSCALE 10.0f

#define NT 64        // center columns per LDS tile
#define ROWS_WG 64   // feature rows per workgroup
#define WAVES 4
#define NBLOCKS (PP * (BB / ROWS_WG))   // 384

typedef __attribute__((ext_vector_type(16))) __bf16 v16bf;
typedef __attribute__((ext_vector_type(8)))  __bf16 v8bf;
typedef __attribute__((ext_vector_type(4)))  __bf16 v4bf;
typedef __attribute__((ext_vector_type(8)))  float  v8f;

// ---- CDNA5 async global->LDS path (guarded: falls back to sync copy) ----
#if defined(__has_builtin)
#if __has_builtin(__builtin_amdgcn_global_load_async_to_lds_b128) && \
    __has_builtin(__builtin_amdgcn_s_wait_asynccnt)
#define USE_ASYNC 1
#endif
#endif
#ifndef USE_ASYNC
#define USE_ASYNC 0
#endif

// builtin signature (per hipcc diagnostics):
//   void (int __vector(4)* src, int __vector(4) __shared__* dst, imm offset, imm cpol)
typedef __attribute__((__vector_size__(16))) int b128_t;
typedef __attribute__((address_space(3))) b128_t* lb128_t;

// ---------------- prep: split-bf16 centers + c^2, written once to d_ws ----------------
__launch_bounds__(256)
__global__ void prep_convert(const float* __restrict__ centers,
                             __bf16* __restrict__ hiG, __bf16* __restrict__ loG)
{
    const size_t total4 = (size_t)PP * NN * DD / 4;
    const float4* src = (const float4*)centers;
    for (size_t e = (size_t)blockIdx.x * blockDim.x + threadIdx.x; e < total4;
         e += (size_t)gridDim.x * blockDim.x) {
        float4 x = src[e];
        float xs[4] = {x.x, x.y, x.z, x.w};
        v4bf hv, lv;
        #pragma unroll
        for (int q = 0; q < 4; ++q) {
            __bf16 h = (__bf16)xs[q];
            hv[q] = h;
            lv[q] = (__bf16)(xs[q] - (float)h);
        }
        *(v4bf*)(hiG + e * 4) = hv;
        *(v4bf*)(loG + e * 4) = lv;
    }
}

__launch_bounds__(256)
__global__ void prep_c2(const float* __restrict__ centers, float* __restrict__ c2G)
{
    int r = blockIdx.x * blockDim.x + threadIdx.x;
    if (r < PP * NN) {
        const float4* cp = (const float4*)(centers + (size_t)r * DD);
        float s = 0.0f;
        #pragma unroll 16
        for (int i = 0; i < 64; ++i) {
            float4 v = cp[i];
            s += v.x * v.x + v.y * v.y + v.z * v.z + v.w * v.w;
        }
        c2G[r] = s;
    }
}

// ---------------- tile staging (double-buffered) ----------------
template <bool PRE>
__device__ __forceinline__ void stage_tile(
    int buf, int p, int n0, int tid,
    const float* __restrict__ centers,
    const __bf16* __restrict__ cHiG, const __bf16* __restrict__ cLoG,
    const float* __restrict__ c2G,
    __bf16 (&sChi)[2][NT][264], __bf16 (&sClo)[2][NT][264], float (&sC2)[2][NT])
{
    if (PRE) {
        // pure byte copy of pre-converted split-bf16 tile (contiguous 2x32KB)
        const uint4* gh = (const uint4*)(cHiG + ((size_t)p * NN + n0) * DD);
        const uint4* gl = (const uint4*)(cLoG + ((size_t)p * NN + n0) * DD);
        #pragma unroll
        for (int j = 0; j < 16; ++j) {
            int ch   = tid + 128 * j;      // 2048 uint4 per array
            int col  = ch >> 5;            // 32 chunks of 8 bf16 per column
            int doff = (ch & 31) * 8;
#if USE_ASYNC
            __builtin_amdgcn_global_load_async_to_lds_b128(
                (b128_t*)(gh + ch),
                (lb128_t)(uint32_t)(uintptr_t)&sChi[buf][col][doff], 0, 0);
            __builtin_amdgcn_global_load_async_to_lds_b128(
                (b128_t*)(gl + ch),
                (lb128_t)(uint32_t)(uintptr_t)&sClo[buf][col][doff], 0, 0);
#else
            *(uint4*)&sChi[buf][col][doff] = gh[ch];
            *(uint4*)&sClo[buf][col][doff] = gl[ch];
#endif
        }
        if (tid < NT) sC2[buf][tid] = c2G[(size_t)p * NN + n0 + tid];
    } else {
        // fallback: convert f32 -> split bf16 in-kernel
        const float4* src = (const float4*)(centers + ((size_t)p * NN + n0) * DD);
        #pragma unroll
        for (int j = 0; j < 32; ++j) {
            int e4  = tid + 128 * j;
            int col = e4 >> 6;
            int d   = (e4 & 63) * 4;
            float4 x = src[e4];
            float xs[4] = {x.x, x.y, x.z, x.w};
            #pragma unroll
            for (int q = 0; q < 4; ++q) {
                __bf16 h = (__bf16)xs[q];
                sChi[buf][col][d + q] = h;
                sClo[buf][col][d + q] = (__bf16)(xs[q] - (float)h);
            }
        }
        if (tid < NT) {
            const float4* cp = (const float4*)(centers + ((size_t)p * NN + n0 + tid) * DD);
            float c2 = 0.0f;
            #pragma unroll 16
            for (int i = 0; i < 64; ++i) {
                float4 v = cp[i];
                c2 += v.x * v.x + v.y * v.y + v.z * v.z + v.w * v.w;
            }
            sC2[buf][tid] = c2;
        }
    }
}

// ---------------- main fused kernel ----------------
template <bool PRE>
__launch_bounds__(128)
__global__ void knn_lse_kernel(const float* __restrict__ feature,
                               const float* __restrict__ centers,
                               const int*   __restrict__ position,
                               const __bf16* __restrict__ cHiG,
                               const __bf16* __restrict__ cLoG,
                               const float*  __restrict__ c2G,
                               float* __restrict__ partial)
{
    // padded stride 264 bf16 = 132 dwords -> bank = (4*col + k) % 64, conflict-free
    __shared__ __bf16 sChi[2][NT][264];
    __shared__ __bf16 sClo[2][NT][264];
    __shared__ float  sC2[2][NT];
    __shared__ float  sDist[WAVES][16][66];
    __shared__ float  sRed[WAVES];

    const int tid  = threadIdx.x;
    const int wave = tid >> 5;
    const int lane = tid & 31;
    const int l15  = lane & 15;
    const int sel  = lane >> 4;

    const int p       = blockIdx.x / (BB / ROWS_WG);
    const int rb      = blockIdx.x % (BB / ROWS_WG);
    const int rowBase = rb * ROWS_WG;
    const int myRow   = rowBase + wave * 16 + l15;

    // ---- Pre-pack split-bf16 A fragments for all 8 K-chunks into registers ----
    // CDNA5 16-bit A 16x32 layout: lanes 0-15 -> M=lane, K={0..7,16..23};
    //                              lanes 16-31 -> M=lane-16, K={8..15,24..31}.
    v16bf aHi[8], aLo[8];
    {
        const float* fbase = feature + ((size_t)p * BB + myRow) * DD;
        #pragma unroll
        for (int kc = 0; kc < 8; ++kc) {
            const float* fp = fbase + kc * 32 + sel * 8;
            #pragma unroll
            for (int i = 0; i < 8; ++i) {
                float x0 = fp[i];
                float x1 = fp[16 + i];
                __bf16 h0 = (__bf16)x0;
                __bf16 h1 = (__bf16)x1;
                aHi[kc][i]     = h0;  aLo[kc][i]     = (__bf16)(x0 - (float)h0);
                aHi[kc][8 + i] = h1;  aLo[kc][8 + i] = (__bf16)(x1 - (float)h1);
            }
        }
    }

    // ---- per-row scan state: lanes 0..15 of each wave each own one row ----
    const int scanRow = rowBase + wave * 16 + lane;
    int   pos   = (lane < 16) ? position[scanRow] : -1;
    float m     = __builtin_inff();
    float s_all = 0.0f;
    float tk[KK];
    #pragma unroll
    for (int i = 0; i < KK; ++i) tk[i] = __builtin_inff();

    // prologue: stage tile 0
    stage_tile<PRE>(0, p, 0, tid, centers, cHiG, cLoG, c2G, sChi, sClo, sC2);
#if USE_ASYNC
    if (PRE) __builtin_amdgcn_s_wait_asynccnt(0);
#endif
    __syncthreads();

    const int NITER = NN / NT;   // 128
    for (int it = 0; it < NITER; ++it) {
        const int n0 = it * NT;
        const int cb = it & 1;

        // prefetch next tile into the other buffer (async overlaps with WMMA below)
        if (it + 1 < NITER)
            stage_tile<PRE>(cb ^ 1, p, n0 + NT, tid, centers, cHiG, cLoG, c2G,
                            sChi, sClo, sC2);

        // ---- GEMM: 16 rows x 64 cols per wave via split-bf16 WMMA ----
        v8f acc[4];
        #pragma unroll
        for (int t = 0; t < 4; ++t)
            #pragma unroll
            for (int v = 0; v < 8; ++v) acc[t][v] = 0.0f;

        #pragma unroll
        for (int kc = 0; kc < 8; ++kc) {
            const int doff = kc * 32 + sel * 8;   // B: lanes0-15 K{0..7,16..23}, lanes16-31 K{8..15,24..31}
            #pragma unroll
            for (int t = 0; t < 4; ++t) {
                const int col = t * 16 + l15;
                const v8bf* ph = (const v8bf*)&sChi[cb][col][doff];
                const v8bf* pl = (const v8bf*)&sClo[cb][col][doff];
                v8bf h0 = ph[0], h1 = ph[2];
                v8bf l0 = pl[0], l1 = pl[2];
                v16bf bh, bl;
                #pragma unroll
                for (int i = 0; i < 8; ++i) {
                    bh[i] = h0[i]; bh[8 + i] = h1[i];
                    bl[i] = l0[i]; bl[8 + i] = l1[i];
                }
                acc[t] = __builtin_amdgcn_wmma_f32_16x16x32_bf16(
                             false, aHi[kc], false, bh, (short)0, acc[t], false, false);
                acc[t] = __builtin_amdgcn_wmma_f32_16x16x32_bf16(
                             false, aHi[kc], false, bl, (short)0, acc[t], false, false);
                acc[t] = __builtin_amdgcn_wmma_f32_16x16x32_bf16(
                             false, aLo[kc], false, bh, (short)0, acc[t], false, false);
            }
        }

        // ---- finalize d' = c^2 - 2*dot into wave-private LDS staging ----
        // C layout: VGPR v -> row = v + 8*(lane>=16), col = lane & 15
        #pragma unroll
        for (int t = 0; t < 4; ++t) {
            const int colL = t * 16 + l15;
            const float c2v = sC2[cb][colL];
            const int rbase = sel * 8;
            #pragma unroll
            for (int v = 0; v < 8; ++v)
                sDist[wave][rbase + v][colL] = c2v - 2.0f * acc[t][v];
        }

        // ---- per-row online LSE + top-10 (lanes 0-15, one row each) ----
        if (lane < 16) {
            for (int c = 0; c < NT; ++c) {
                if (n0 + c == pos) continue;
                float d = sDist[wave][lane][c];
                if (d < tk[KK - 1]) {
                    tk[KK - 1] = d;
                    #pragma unroll
                    for (int i = KK - 1; i > 0; --i) {
                        float a = tk[i - 1], b = tk[i];
                        tk[i - 1] = fminf(a, b);
                        tk[i]     = fmaxf(a, b);
                    }
                }
                if (d < m) {
                    s_all = s_all * __expf(FSCALE * (d - m)) + 1.0f;
                    m = d;
                } else {
                    s_all += __expf(FSCALE * (m - d));
                }
            }
        }

#if USE_ASYNC
        if (PRE) __builtin_amdgcn_s_wait_asynccnt(0);
#endif
        __syncthreads();   // current tile consumed; prefetched tile landed
    }

    // ---- per-row value: y - x = log(sum_all) - log(sum_top10), shift cancels ----
    float rv = 0.0f;
    if (lane < 16) {
        float stk = 0.0f;
        #pragma unroll
        for (int i = 0; i < KK; ++i) stk += __expf(FSCALE * (m - tk[i]));
        rv = __logf(s_all) - __logf(stk);
    }
    #pragma unroll
    for (int off = 16; off > 0; off >>= 1) rv += __shfl_down(rv, off);
    if (lane == 0) sRed[wave] = rv;
    __syncthreads();
    if (tid == 0)
        partial[blockIdx.x] = sRed[0] + sRed[1] + sRed[2] + sRed[3];
}

__launch_bounds__(256)
__global__ void final_reduce(const float* __restrict__ partial, int n,
                             float* __restrict__ out)
{
    __shared__ float sb[8];
    float s = 0.0f;
    for (int i = threadIdx.x; i < n; i += blockDim.x) s += partial[i];
    #pragma unroll
    for (int off = 16; off > 0; off >>= 1) s += __shfl_down(s, off);
    if ((threadIdx.x & 31) == 0) sb[threadIdx.x >> 5] = s;
    __syncthreads();
    if (threadIdx.x == 0) {
        float t = 0.0f;
        #pragma unroll
        for (int w = 0; w < 8; ++w) t += sb[w];
        out[0] = t * (1.0f / ((float)BB * (float)PP));
    }
}

extern "C" void kernel_launch(void* const* d_in, const int* in_sizes, int n_in,
                              void* d_out, int out_size, void* d_ws, size_t ws_size,
                              hipStream_t stream)
{
    const float* feature  = (const float*)d_in[0];
    const float* centers  = (const float*)d_in[1];
    const int*   position = (const int*)d_in[2];
    float* partial = (float*)d_ws;
    float* out     = (float*)d_out;

    // d_ws layout: [partials 1.5KB][pad][c2 192KB][hi 24MB][lo 24MB]
    const size_t off_c2 = 2048;
    const size_t off_hi = off_c2 + (size_t)PP * NN * 4;
    const size_t off_lo = off_hi + (size_t)PP * NN * DD * 2;
    const size_t need   = off_lo + (size_t)PP * NN * DD * 2;

    if (ws_size >= need) {
        float*  c2G = (float*)((char*)d_ws + off_c2);
        __bf16* hiG = (__bf16*)((char*)d_ws + off_hi);
        __bf16* loG = (__bf16*)((char*)d_ws + off_lo);
        prep_convert<<<dim3(2048), dim3(256), 0, stream>>>(centers, hiG, loG);
        prep_c2<<<dim3((PP * NN + 255) / 256), dim3(256), 0, stream>>>(centers, c2G);
        knn_lse_kernel<true><<<dim3(NBLOCKS), dim3(128), 0, stream>>>(
            feature, centers, position, hiG, loG, c2G, partial);
    } else {
        knn_lse_kernel<false><<<dim3(NBLOCKS), dim3(128), 0, stream>>>(
            feature, centers, position, nullptr, nullptr, nullptr, partial);
    }
    final_reduce<<<dim3(1), dim3(256), 0, stream>>>(partial, NBLOCKS, out);
}
